// ModelWithEmbeddingIB_9612136809124
// MI455X (gfx1250) — matrix operand order repaired and checked by reading, not compiled, not run on hardware
//
#include <hip/hip_runtime.h>
#include <stdint.h>

// ---------------- problem constants ----------------
#define B_ROWS 8192
#define D_DIM  1024
#define H_DIM  2048
#define Z_DIM  512
#define C_CLS  32
#define LEAKK  0.01f

typedef __attribute__((ext_vector_type(16))) __bf16 v16bf;
typedef __attribute__((ext_vector_type(8)))  float  v8f;
typedef __attribute__((ext_vector_type(4)))  unsigned int v4u32;
typedef __attribute__((ext_vector_type(8)))  int v8i32;
typedef __attribute__((ext_vector_type(4)))  int v4i32;

__device__ __forceinline__ unsigned short f2bf(float f) {
  unsigned int u = __float_as_uint(f);
  unsigned int r = (u + 0x7FFFu + ((u >> 16) & 1u)) >> 16;  // RNE
  return (unsigned short)r;
}

// ---------------- TDM: 2D bf16 tile -> LDS (padded rows) ----------------
// D# per CDNA5 ISA ch.8: group0 = {flags, lds_addr, global_addr, type},
// group1 = {mask/data_size/pad, tensor dims, tile dims, strides}. Dims are
// given relative to the tile start so TDM's OOB zero-fill handles N edges.
// Pad: every 16 DWORDs (64B row = 32 halfs) insert 4 DWORDs (16B) -> LDS row
// stride 80B = LDT halfs, matching the fragment-read addressing below.
__device__ __forceinline__ void tdm_load_tile_bf16(
    unsigned lds_addr, const unsigned short* gsrc,
    int cols_rem, int rows_rem, int tile_cols, int tile_rows,
    long long stride_elems) {
  unsigned long long ga = (unsigned long long)(size_t)gsrc;
  v4u32 g0;
  g0[0] = 1u;                                        // count=1, user descriptor
  g0[1] = lds_addr;                                  // LDS byte address
  g0[2] = (unsigned)(ga & 0xFFFFFFFFu);              // global_addr[31:0]
  g0[3] = ((unsigned)(ga >> 32) & 0x01FFFFFFu)       // global_addr[56:32]
          | 0x80000000u;                             // type = 2 ("image")
  v8i32 g1;
  g1[0] = (int)((1u << 16)       // data_size = 2 bytes
              | (1u << 20)       // pad_enable
              | (3u << 22)       // pad_interval: 16 DWORDs
              | (3u << 25));     // pad_amount:   4 DWORDs
  g1[1] = (int)(((unsigned)cols_rem & 0xFFFFu) << 16);              // dim0[15:0]
  g1[2] = (int)((((unsigned)cols_rem >> 16) & 0xFFFFu)
              | (((unsigned)rows_rem & 0xFFFFu) << 16));            // dim0 hi | dim1 lo
  g1[3] = (int)((((unsigned)rows_rem >> 16) & 0xFFFFu)
              | ((unsigned)tile_cols << 16));                       // dim1 hi | tile_dim0
  g1[4] = (int)((unsigned)tile_rows);                               // tile_dim1 (tile_dim2=0)
  unsigned long long st = (unsigned long long)stride_elems;
  g1[5] = (int)(st & 0xFFFFFFFFu);                                  // dim0_stride lo
  g1[6] = (int)((st >> 32) & 0xFFFFu);                              // dim0_stride hi
  g1[7] = 0;                                                        // dim1_stride = 0
  v4i32 z4 = {0, 0, 0, 0};
  v8i32 z8 = {0, 0, 0, 0, 0, 0, 0, 0};
  __builtin_amdgcn_tensor_load_to_lds(g0, g1, z4, z4, z8, 0);
}

// ---------------- WMMA GEMM: C[M,N] = A[M,K](bf16) @ Bt[N,K](bf16) + bias ----------------
#define BM 128
#define BN 128
#define BK 32
#define LDT 40  // padded LDS row stride (halfs); 80B = 5*16B

union FragAB { v16bf v; uint4 q[2]; };

__global__ void __launch_bounds__(256)
gemm_bf16_wmma(const unsigned short* __restrict__ A,
               const unsigned short* __restrict__ Bt,
               const float* __restrict__ bias,
               float* __restrict__ Cf, unsigned short* __restrict__ Cb,
               int M, int N, int K, int do_leaky) {
  __shared__ unsigned short sA[2][BM * LDT];
  __shared__ unsigned short sB[2][BN * LDT];

  const int t    = threadIdx.x;
  const int lane = t & 31;
  const int wave = t >> 5;
  const int g    = lane >> 4;   // half-wave group (0/1)
  const int ln   = lane & 15;
  const int wm   = wave & 1;    // 2 wave-rows  (64 rows each)
  const int wn   = wave >> 1;   // 4 wave-cols  (32 cols each)
  const int m0   = blockIdx.y * BM;
  const int n0   = blockIdx.x * BN;

  v8f acc[4][2] = {};

  // prologue: DMA first K tile into buffer 0 (wave 0 -> A, wave 1 -> B)
  if (wave == 0)
    tdm_load_tile_bf16((unsigned)(size_t)&sA[0][0], A + (size_t)m0 * K,
                       K, M - m0, BK, BM, K);
  else if (wave == 1)
    tdm_load_tile_bf16((unsigned)(size_t)&sB[0][0], Bt + (size_t)n0 * K,
                       K, N - n0, BK, BN, K);

  int cur = 0;
  for (int kt = 0; kt < K; kt += BK) {
    // issue DMA for next tile into the other buffer, then retire current tile
    if (kt + BK < K) {
      if (wave == 0)
        tdm_load_tile_bf16((unsigned)(size_t)&sA[cur ^ 1][0],
                           A + (size_t)m0 * K + (kt + BK),
                           K - (kt + BK), M - m0, BK, BM, K);
      else if (wave == 1)
        tdm_load_tile_bf16((unsigned)(size_t)&sB[cur ^ 1][0],
                           Bt + (size_t)n0 * K + (kt + BK),
                           K - (kt + BK), N - n0, BK, BN, K);
      if (wave < 2) __builtin_amdgcn_s_wait_tensorcnt(1);  // oldest DMA (current) done
    } else {
      if (wave < 2) __builtin_amdgcn_s_wait_tensorcnt(0);
    }
    __syncthreads();  // current tile visible to all waves

    // ---- fragment loads per ISA 16-bit layouts ----
    FragAB fa[4], fb[2];
#pragma unroll
    for (int nf = 0; nf < 2; ++nf) {
      const unsigned short* p = &sB[cur][(wn * 32 + nf * 16 + ln) * LDT];
      fb[nf].q[0] = *(const uint4*)(p + g * 8);        // K = g*8 .. +7
      fb[nf].q[1] = *(const uint4*)(p + 16 + g * 8);   // K = 16+g*8 .. +7
    }
#pragma unroll
    for (int mf = 0; mf < 4; ++mf) {
      const unsigned short* p = &sA[cur][(wm * 64 + mf * 16 + ln) * LDT];
      fa[mf].q[0] = *(const uint4*)(p + g * 8);        // K = g*8 .. +7
      fa[mf].q[1] = *(const uint4*)(p + 16 + g * 8);   // K = 16+g*8 .. +7
    }
#pragma unroll
    for (int mf = 0; mf < 4; ++mf)
#pragma unroll
      for (int nf = 0; nf < 2; ++nf)
        acc[mf][nf] = __builtin_amdgcn_wmma_f32_16x16x32_bf16(
            false, fa[mf].v, false, fb[nf].v, (short)0, acc[mf][nf], false, false);

    __syncthreads();  // all reads done before buffer is overwritten next round
    cur ^= 1;
  }

  // ---- epilogue: C VGPR r -> row (r + 8*g), col = ln ----
#pragma unroll
  for (int mf = 0; mf < 4; ++mf)
#pragma unroll
    for (int nf = 0; nf < 2; ++nf) {
      int col = n0 + wn * 32 + nf * 16 + ln;
      if (col < N) {
        float bv = bias ? bias[col] : 0.0f;
#pragma unroll
        for (int r = 0; r < 8; ++r) {
          int row = m0 + wm * 64 + mf * 16 + g * 8 + r;
          float v = acc[mf][nf][r] + bv;
          if (do_leaky) v = (v >= 0.0f) ? v : LEAKK * v;
          if (Cf) Cf[(size_t)row * N + col] = v;
          if (Cb) Cb[(size_t)row * N + col] = f2bf(v);
        }
      }
    }
}

// ---------------- elementwise / conversion kernels ----------------
__global__ void f32_to_bf16_kernel(const float* __restrict__ src,
                                   unsigned short* __restrict__ dst, size_t n) {
  size_t i = (size_t)blockIdx.x * blockDim.x + threadIdx.x;
  if (i < n) dst[i] = f2bf(src[i]);
}

// dst[N][K] = bf16(src[K][N])
__global__ void transpose_bf16_kernel(const float* __restrict__ src,
                                      unsigned short* __restrict__ dst, int K, int N) {
  size_t i = (size_t)blockIdx.x * blockDim.x + threadIdx.x;
  if (i < (size_t)K * N) {
    int n = (int)(i / K), k = (int)(i % K);
    dst[i] = f2bf(src[(size_t)k * N + n]);
  }
}

__global__ void gather_embed_kernel(const float* __restrict__ E,
                                    const int* __restrict__ labels,
                                    unsigned short* __restrict__ dst) {
  size_t i = (size_t)blockIdx.x * blockDim.x + threadIdx.x;
  if (i < (size_t)B_ROWS * D_DIM) {
    int b = (int)(i / D_DIM), d = (int)(i % D_DIM);
    dst[i] = f2bf(E[(size_t)labels[b] * D_DIM + d]);
  }
}

// ---------------- deterministic counting sort (order/perm) ----------------
__global__ void class_count_kernel(const int* __restrict__ labels, int* __restrict__ counts) {
  int c = blockIdx.x, lane = threadIdx.x;  // blockDim = 32 (one wave)
  int cnt = 0;
  for (int base = 0; base < B_ROWS; base += 32) {
    unsigned long long mask = __ballot(labels[base + lane] == c);
    cnt += __popcll(mask);
  }
  if (lane == 0) counts[c] = cnt;
}

__global__ void class_offsets_kernel(const int* __restrict__ counts, int* __restrict__ offsets) {
  if (threadIdx.x == 0 && blockIdx.x == 0) {
    int run = 0;
    for (int c = 0; c < C_CLS; ++c) { offsets[c] = run; run += counts[c]; }
  }
}

__global__ void class_scatter_kernel(const int* __restrict__ labels,
                                     const int* __restrict__ counts,
                                     const int* __restrict__ offsets,
                                     int* __restrict__ order, int* __restrict__ perm) {
  int c = blockIdx.x, lane = threadIdx.x;  // blockDim = 32
  int off = offsets[c], cnt = counts[c], run = 0;
  for (int base = 0; base < B_ROWS; base += 32) {
    int i = base + lane;
    bool m = (labels[i] == c);
    unsigned long long mask = __ballot(m);
    int rank = __popcll(mask & ((1ull << lane) - 1ull));
    if (m) {
      order[off + run + rank] = i;                      // stable within group
      perm[off + (cnt - 1) - (run + rank)] = i;         // deterministic within-group shuffle
    }
    run += __popcll(mask);
  }
}

// ---------------- cosine / contrastive / OT ----------------
__global__ void __launch_bounds__(256)
cos_stats_kernel(const float* __restrict__ zl, const float* __restrict__ zr,
                 const float* __restrict__ zi, float* __restrict__ cost,
                 float* __restrict__ ps) {
  __shared__ float red[5][256];
  int b = blockIdx.x;
  const float* pl = zl + (size_t)b * Z_DIM;
  const float* pr = zr + (size_t)b * Z_DIM;
  const float* pi = zi + (size_t)b * Z_DIM;
  float ll = 0, rr = 0, ii = 0, lr = 0, li = 0;
  for (int j = threadIdx.x; j < Z_DIM; j += 256) {
    float a = pl[j], c2 = pr[j], d = pi[j];
    ll += a * a; rr += c2 * c2; ii += d * d; lr += a * c2; li += a * d;
  }
  red[0][threadIdx.x] = ll; red[1][threadIdx.x] = rr; red[2][threadIdx.x] = ii;
  red[3][threadIdx.x] = lr; red[4][threadIdx.x] = li;
  __syncthreads();
  for (int s = 128; s > 0; s >>= 1) {
    if (threadIdx.x < s)
      for (int q = 0; q < 5; ++q) red[q][threadIdx.x] += red[q][threadIdx.x + s];
    __syncthreads();
  }
  if (threadIdx.x == 0) {
    ll = red[0][0]; rr = red[1][0]; ii = red[2][0]; lr = red[3][0]; li = red[4][0];
    float nl5 = fmaxf(sqrtf(ll), 1e-5f), nr5 = fmaxf(sqrtf(rr), 1e-5f);
    cost[b] = 1.0f - lr / (nl5 * nr5);
    float nl = fmaxf(sqrtf(ll), 1e-8f), nr = fmaxf(sqrtf(rr), 1e-8f), ni = fmaxf(sqrtf(ii), 1e-8f);
    float m = 1.5f * (li / (nl * ni) - lr / (nl * nr));
    ps[b] = fmaxf(m, 0.0f) + log1pf(expf(-fabsf(m)));  // softplus = -log_softmax[:,0]
  }
}

// IPOT with sigma=T=1 is a fixed point (T stays 1) -> ot = mean(cost)
__global__ void __launch_bounds__(1024)
mean_kernel(const float* __restrict__ v, float* __restrict__ out, int n) {
  __shared__ float red[1024];
  float s = 0;
  for (int i = threadIdx.x; i < n; i += 1024) s += v[i];
  red[threadIdx.x] = s;
  __syncthreads();
  for (int st = 512; st > 0; st >>= 1) {
    if (threadIdx.x < st) red[threadIdx.x] += red[threadIdx.x + st];
    __syncthreads();
  }
  if (threadIdx.x == 0) out[0] = red[0] / (float)n;
}

__global__ void __launch_bounds__(256)
group_sum_kernel(const float* __restrict__ ps, const int* __restrict__ labels,
                 float* __restrict__ gsum, float* __restrict__ gcnt) {
  __shared__ float rs[256], rc[256];
  int c = blockIdx.x;
  float s = 0, k = 0;
  for (int i = threadIdx.x; i < B_ROWS; i += 256)
    if (labels[i] == c) { s += ps[i]; k += 1.0f; }
  rs[threadIdx.x] = s; rc[threadIdx.x] = k;
  __syncthreads();
  for (int st = 128; st > 0; st >>= 1) {
    if (threadIdx.x < st) { rs[threadIdx.x] += rs[threadIdx.x + st]; rc[threadIdx.x] += rc[threadIdx.x + st]; }
    __syncthreads();
  }
  if (threadIdx.x == 0) { gsum[c] = rs[0]; gcnt[c] = rc[0]; }
}

__global__ void contrastive_kernel(const float* __restrict__ gsum,
                                   const float* __restrict__ gcnt, float* __restrict__ out) {
  if (threadIdx.x == 0 && blockIdx.x == 0) {
    float tot = 0, np = 0;
    for (int c = 0; c < C_CLS; ++c)
      if (gcnt[c] > 0.0f) { tot += gsum[c] / gcnt[c]; np += 1.0f; }
    out[0] = (np > 0.0f) ? tot / np : 0.0f;
  }
}

// ---------------- BatchNorm (batch stats) ----------------
__global__ void __launch_bounds__(256)
bn_stats_kernel(const float* __restrict__ hc, float* __restrict__ mu, float* __restrict__ rstd) {
  __shared__ float r1[256], r2[256];
  int j = blockIdx.x;
  float s = 0, s2 = 0;
  for (int i = threadIdx.x; i < B_ROWS; i += 256) {
    float v = hc[(size_t)i * H_DIM + j];
    s += v; s2 += v * v;
  }
  r1[threadIdx.x] = s; r2[threadIdx.x] = s2;
  __syncthreads();
  for (int st = 128; st > 0; st >>= 1) {
    if (threadIdx.x < st) { r1[threadIdx.x] += r1[threadIdx.x + st]; r2[threadIdx.x] += r2[threadIdx.x + st]; }
    __syncthreads();
  }
  if (threadIdx.x == 0) {
    float m = r1[0] / (float)B_ROWS;
    float var = r2[0] / (float)B_ROWS - m * m;
    mu[j] = m;
    rstd[j] = rsqrtf(var + 1e-5f);
  }
}

__global__ void bn_apply_kernel(const float* __restrict__ hc, const float* __restrict__ mu,
                                const float* __restrict__ rstd, const float* __restrict__ gamma,
                                const float* __restrict__ beta, unsigned short* __restrict__ dst) {
  size_t i = (size_t)blockIdx.x * blockDim.x + threadIdx.x;
  if (i < (size_t)B_ROWS * H_DIM) {
    int j = (int)(i % H_DIM);
    float v = (hc[i] - mu[j]) * rstd[j] * gamma[j] + beta[j];
    v = (v >= 0.0f) ? v : LEAKK * v;  // leaky before classifier head
    dst[i] = f2bf(v);
  }
}

// ---------------- gathers ----------------
__global__ void decin_kernel(const unsigned short* __restrict__ zrb,
                             const unsigned short* __restrict__ zib,
                             const int* __restrict__ perm, const int* __restrict__ order,
                             unsigned short* __restrict__ dst) {
  size_t i = (size_t)blockIdx.x * blockDim.x + threadIdx.x;
  if (i < (size_t)B_ROWS * 2 * Z_DIM) {
    int b = (int)(i / (2 * Z_DIM)), j = (int)(i % (2 * Z_DIM));
    dst[i] = (j < Z_DIM) ? zrb[(size_t)perm[b] * Z_DIM + j]
                         : zib[(size_t)order[b] * Z_DIM + (j - Z_DIM)];
  }
}

__global__ void xori_kernel(const float* __restrict__ x, const int* __restrict__ order,
                            float* __restrict__ dst) {
  size_t i = (size_t)blockIdx.x * blockDim.x + threadIdx.x;
  if (i < (size_t)B_ROWS * D_DIM) {
    int b = (int)(i / D_DIM), j = (int)(i % D_DIM);
    dst[i] = x[(size_t)order[b] * D_DIM + j];
  }
}

// ---------------- host orchestration ----------------
extern "C" void kernel_launch(void* const* d_in, const int* in_sizes, int n_in,
                              void* d_out, int out_size, void* d_ws, size_t ws_size,
                              hipStream_t stream) {
  (void)in_sizes; (void)n_in; (void)out_size; (void)ws_size;
  const float* x      = (const float*)d_in[0];
  const int*   labels = (const int*)d_in[1];
  const float* Wr1 = (const float*)d_in[2];
  const float* br1 = (const float*)d_in[3];
  const float* Wr2 = (const float*)d_in[4];
  const float* br2 = (const float*)d_in[5];
  const float* Wi1 = (const float*)d_in[6];
  const float* bi1 = (const float*)d_in[7];
  const float* Wi2 = (const float*)d_in[8];
  const float* bi2 = (const float*)d_in[9];
  const float* E   = (const float*)d_in[10];
  const float* Wd1 = (const float*)d_in[11];
  const float* bd1 = (const float*)d_in[12];
  const float* Wd2 = (const float*)d_in[13];
  const float* bd2 = (const float*)d_in[14];
  const float* Wc1 = (const float*)d_in[15];
  const float* bc1 = (const float*)d_in[16];
  const float* bng = (const float*)d_in[17];
  const float* bnb = (const float*)d_in[18];
  const float* Wc2 = (const float*)d_in[19];
  const float* bc2 = (const float*)d_in[20];
  float* dout = (float*)d_out;

  // output layout (flat, reference return order)
  const size_t OFF_XREC = 0;
  const size_t OFF_ZL   = (size_t)B_ROWS * D_DIM;
  const size_t OFF_ZR   = OFF_ZL + (size_t)B_ROWS * Z_DIM;
  const size_t OFF_ZI   = OFF_ZR + (size_t)B_ROWS * Z_DIM;
  const size_t OFF_OT   = OFF_ZI + (size_t)B_ROWS * Z_DIM;
  const size_t OFF_XORI = OFF_OT + 1;
  const size_t OFF_CON  = OFF_XORI + (size_t)B_ROWS * D_DIM;
  const size_t OFF_LOG  = OFF_CON + 1;

  // workspace carving
  char* wp = (char*)d_ws;
  auto carve = [&](size_t bytes) -> void* {
    void* p = (void*)wp;
    wp += (bytes + 255) & ~(size_t)255;
    return p;
  };
  unsigned short* x_bf   = (unsigned short*)carve((size_t)B_ROWS * D_DIM * 2);
  unsigned short* xe_bf  = (unsigned short*)carve((size_t)B_ROWS * D_DIM * 2);
  unsigned short* h_bf   = (unsigned short*)carve((size_t)B_ROWS * H_DIM * 2);  // reused
  unsigned short* zr_bf  = (unsigned short*)carve((size_t)B_ROWS * Z_DIM * 2);
  unsigned short* zi_bf  = (unsigned short*)carve((size_t)B_ROWS * Z_DIM * 2);
  unsigned short* di_bf  = (unsigned short*)carve((size_t)B_ROWS * 2 * Z_DIM * 2);
  float*          hc_f   = (float*)carve((size_t)B_ROWS * H_DIM * 4);
  unsigned short* wr1t = (unsigned short*)carve((size_t)D_DIM * H_DIM * 2);
  unsigned short* wr2t = (unsigned short*)carve((size_t)H_DIM * Z_DIM * 2);
  unsigned short* wi1t = (unsigned short*)carve((size_t)D_DIM * H_DIM * 2);
  unsigned short* wi2t = (unsigned short*)carve((size_t)H_DIM * Z_DIM * 2);
  unsigned short* wd1t = (unsigned short*)carve((size_t)(2 * Z_DIM) * H_DIM * 2);
  unsigned short* wd2t = (unsigned short*)carve((size_t)H_DIM * D_DIM * 2);
  unsigned short* wc1t = (unsigned short*)carve((size_t)Z_DIM * H_DIM * 2);
  unsigned short* wc2t = (unsigned short*)carve((size_t)H_DIM * C_CLS * 2);
  int*   order   = (int*)carve((size_t)B_ROWS * 4);
  int*   perm    = (int*)carve((size_t)B_ROWS * 4);
  int*   counts  = (int*)carve(C_CLS * 4);
  int*   offsets = (int*)carve(C_CLS * 4);
  float* cost    = (float*)carve((size_t)B_ROWS * 4);
  float* ps      = (float*)carve((size_t)B_ROWS * 4);
  float* gsum    = (float*)carve(C_CLS * 4);
  float* gcnt    = (float*)carve(C_CLS * 4);
  float* mu      = (float*)carve(H_DIM * 4);
  float* rstd    = (float*)carve(H_DIM * 4);

  auto nb = [](size_t n) { return (unsigned)((n + 255) / 256); };
  auto gemm = [&](const unsigned short* A, const unsigned short* Bt, const float* bias,
                  float* Cf, unsigned short* Cb, int N, int K, int lk) {
    dim3 grid((unsigned)((N + BN - 1) / BN), (unsigned)(B_ROWS / BM));
    gemm_bf16_wmma<<<grid, 256, 0, stream>>>(A, Bt, bias, Cf, Cb, B_ROWS, N, K, lk);
  };

  // ---- conversions / gathers ----
  f32_to_bf16_kernel<<<nb((size_t)B_ROWS * D_DIM), 256, 0, stream>>>(x, x_bf, (size_t)B_ROWS * D_DIM);
  gather_embed_kernel<<<nb((size_t)B_ROWS * D_DIM), 256, 0, stream>>>(E, labels, xe_bf);
  transpose_bf16_kernel<<<nb((size_t)D_DIM * H_DIM), 256, 0, stream>>>(Wr1, wr1t, D_DIM, H_DIM);
  transpose_bf16_kernel<<<nb((size_t)H_DIM * Z_DIM), 256, 0, stream>>>(Wr2, wr2t, H_DIM, Z_DIM);
  transpose_bf16_kernel<<<nb((size_t)D_DIM * H_DIM), 256, 0, stream>>>(Wi1, wi1t, D_DIM, H_DIM);
  transpose_bf16_kernel<<<nb((size_t)H_DIM * Z_DIM), 256, 0, stream>>>(Wi2, wi2t, H_DIM, Z_DIM);
  transpose_bf16_kernel<<<nb((size_t)2 * Z_DIM * H_DIM), 256, 0, stream>>>(Wd1, wd1t, 2 * Z_DIM, H_DIM);
  transpose_bf16_kernel<<<nb((size_t)H_DIM * D_DIM), 256, 0, stream>>>(Wd2, wd2t, H_DIM, D_DIM);
  transpose_bf16_kernel<<<nb((size_t)Z_DIM * H_DIM), 256, 0, stream>>>(Wc1, wc1t, Z_DIM, H_DIM);
  transpose_bf16_kernel<<<nb((size_t)H_DIM * C_CLS), 256, 0, stream>>>(Wc2, wc2t, H_DIM, C_CLS);

  // ---- order / perm (deterministic counting sort over 32 classes) ----
  class_count_kernel<<<C_CLS, 32, 0, stream>>>(labels, counts);
  class_offsets_kernel<<<1, 1, 0, stream>>>(counts, offsets);
  class_scatter_kernel<<<C_CLS, 32, 0, stream>>>(labels, counts, offsets, order, perm);

  // ---- encoders ----
  gemm(x_bf,  wr1t, br1, nullptr,         h_bf,  H_DIM, D_DIM, 1);
  gemm(h_bf,  wr2t, br2, dout + OFF_ZR,   zr_bf, Z_DIM, H_DIM, 0);
  gemm(x_bf,  wi1t, bi1, nullptr,         h_bf,  H_DIM, D_DIM, 1);
  gemm(h_bf,  wi2t, bi2, dout + OFF_ZI,   zi_bf, Z_DIM, H_DIM, 0);
  gemm(xe_bf, wr1t, br1, nullptr,         h_bf,  H_DIM, D_DIM, 1);
  gemm(h_bf,  wr2t, br2, dout + OFF_ZL,   nullptr, Z_DIM, H_DIM, 0);

  // ---- OT + contrastive (IPOT collapses to T=1 -> ot = mean(cost)) ----
  cos_stats_kernel<<<B_ROWS, 256, 0, stream>>>(dout + OFF_ZL, dout + OFF_ZR, dout + OFF_ZI, cost, ps);
  mean_kernel<<<1, 1024, 0, stream>>>(cost, dout + OFF_OT, B_ROWS);
  group_sum_kernel<<<C_CLS, 256, 0, stream>>>(ps, labels, gsum, gcnt);
  contrastive_kernel<<<1, 32, 0, stream>>>(gsum, gcnt, dout + OFF_CON);

  // ---- decoder ----
  decin_kernel<<<nb((size_t)B_ROWS * 2 * Z_DIM), 256, 0, stream>>>(zr_bf, zi_bf, perm, order, di_bf);
  gemm(di_bf, wd1t, bd1, nullptr,         h_bf,   H_DIM, 2 * Z_DIM, 1);
  gemm(h_bf,  wd2t, bd2, dout + OFF_XREC, nullptr, D_DIM, H_DIM, 0);
  xori_kernel<<<nb((size_t)B_ROWS * D_DIM), 256, 0, stream>>>(x, order, dout + OFF_XORI);

  // ---- classifier (BN with batch stats, then leaky, then head) ----
  gemm(zr_bf, wc1t, bc1, hc_f, nullptr, H_DIM, Z_DIM, 0);
  bn_stats_kernel<<<H_DIM, 256, 0, stream>>>(hc_f, mu, rstd);
  bn_apply_kernel<<<nb((size_t)B_ROWS * H_DIM), 256, 0, stream>>>(hc_f, mu, rstd, bng, bnb, h_bf);
  gemm(h_bf, wc2t, bc2, dout + OFF_LOG, nullptr, C_CLS, H_DIM, 0);
}
